// PyTorchDeltaNetLayer_70205535420782
// MI455X (gfx1250) — matrix operand order repaired and checked by reading, not compiled, not run on hardware
//
#include <hip/hip_runtime.h>

typedef __attribute__((ext_vector_type(16))) _Float16 v16h;
typedef __attribute__((ext_vector_type(8)))  _Float16 v8h;
typedef __attribute__((ext_vector_type(4)))  _Float16 v4h;
typedef __attribute__((ext_vector_type(8)))  float    v8f;
typedef __attribute__((ext_vector_type(4)))  unsigned int v4u;
typedef __attribute__((ext_vector_type(8)))  int      v8i;
typedef __attribute__((ext_vector_type(4)))  int      v4i;

#define B_  4
#define T_  2048
#define D_  1024
#define H_  16
#define HD_ 64
#define M_  (B_ * T_)      // 8192 token rows
#define MIXC 0.05f

// LDS tile geometry: rows of 32 halves (64B) padded by TDM to 80B stride.
#define LDS_ROW   40                    // halves (80 bytes)
#define A_TILE_B  (64  * LDS_ROW * 2)   // 5120 B
#define B_TILE_B  (128 * LDS_ROW * 2)   // 10240 B
#define SMEM_B    (2 * A_TILE_B + 2 * B_TILE_B)   // 30720 B

union V16U { v16h v; v8h h[2]; };

__device__ __forceinline__ float fsigmoid(float x) { return 1.0f / (1.0f + __expf(-x)); }
__device__ __forceinline__ float fsilu(float x)    { return x * fsigmoid(x); }

// ---------------------------------------------------------------------------
// f32 -> f16 cast, vectorized by 4
// ---------------------------------------------------------------------------
__global__ void cast_f32_to_f16(const float* __restrict__ src,
                                _Float16* __restrict__ dst, int n4) {
    int i = blockIdx.x * blockDim.x + threadIdx.x;
    if (i < n4) {
        float4 f = reinterpret_cast<const float4*>(src)[i];
        v4h h;
        h.x = (_Float16)f.x; h.y = (_Float16)f.y;
        h.z = (_Float16)f.z; h.w = (_Float16)f.w;
        reinterpret_cast<v4h*>(dst)[i] = h;
    }
}

// ---------------------------------------------------------------------------
// TDM: 2D tile (tile0 x tile1 elements, 2B each) from row-major tensor
// (ld = tensor_dim0_stride, in elements) into LDS at lds_off, with LDS
// padding of 4 DWORDs every 16 DWORDs (64B row -> 80B stride).
// D# layout per CDNA5 ISA.
// ---------------------------------------------------------------------------
__device__ __forceinline__ void tdm_load_2d(unsigned int lds_off,
                                            const _Float16* gbase,
                                            unsigned long long row0,
                                            unsigned long long k0,
                                            unsigned int ld,
                                            unsigned int tdim0, unsigned int tdim1,
                                            unsigned int tile0, unsigned int tile1) {
    unsigned long long ga =
        (unsigned long long)(uintptr_t)gbase + (row0 * ld + k0) * 2ull;
    v4u g0;
    g0.x = 1u;                                   // count=1, user descriptor
    g0.y = lds_off;                              // lds_addr (bytes)
    g0.z = (unsigned int)ga;                     // global_addr[31:0]
    g0.w = (unsigned int)((ga >> 32) & 0x01FFFFFFu) | (2u << 30);  // addr hi | type=2
    v8i g1;
    g1[0] = (int)((1u << 16)      // data_size = 2 bytes
                | (1u << 20)      // pad_enable
                | (3u << 22)      // pad_interval: every 16 DWORDs (64B)
                | (3u << 25));    // pad_amount: 4 DWORDs (16B)
    g1[1] = (int)((tdim0 & 0xFFFFu) << 16);                        // tensor_dim0 lo
    g1[2] = (int)(((tdim0 >> 16) & 0xFFFFu) | ((tdim1 & 0xFFFFu) << 16));
    g1[3] = (int)(((tdim1 >> 16) & 0xFFFFu) | (tile0 << 16));      // tile_dim0
    g1[4] = (int)tile1;                                            // tile_dim1 (tile_dim2=0)
    g1[5] = (int)ld;                                               // tensor_dim0_stride lo
    g1[6] = 0;
    g1[7] = 0;
    v4i z4 = {0, 0, 0, 0};
#if defined(__clang_major__) && (__clang_major__ >= 23)
    v8i z8 = {0, 0, 0, 0, 0, 0, 0, 0};
    __builtin_amdgcn_tensor_load_to_lds(g0, g1, z4, z4, z8, 0);
#else
    __builtin_amdgcn_tensor_load_to_lds(g0, g1, z4, z4, 0);
#endif
}

// ---------------------------------------------------------------------------
// Fragment loaders.
// A (16-bit 16x32 MxK): lane L holds row M=L&15; VGPR0-3: K=(L<16?0..7:8..15),
//   VGPR4-7: +16.  B (16-bit 32x16 KxN): lane L holds col N=L&15; same K
//   order as A: VGPR0-3 K=(L<16?0..7:8..15), VGPR4-7: +16.
// ---------------------------------------------------------------------------
__device__ __forceinline__ v16h lds_fragA(const _Float16* base, int row, int lane) {
    const _Float16* p = base + row * LDS_ROW + ((lane >> 4) << 3);
    V16U a;
    a.h[0] = *reinterpret_cast<const v8h*>(p);
    a.h[1] = *reinterpret_cast<const v8h*>(p + 16);
    return a.v;
}
__device__ __forceinline__ v16h lds_fragB(const _Float16* base, int row, int lane) {
    const _Float16* p = base + row * LDS_ROW + ((lane >> 4) << 3);
    V16U b;
    b.h[0] = *reinterpret_cast<const v8h*>(p);
    b.h[1] = *reinterpret_cast<const v8h*>(p + 16);
    return b.v;
}
// Direct-global variants (used by the small H-projection kernel)
__device__ __forceinline__ v16h glb_fragA(const _Float16* __restrict__ X,
                                          int ld, int row, int k0, int lane) {
    const _Float16* p = X + (size_t)row * ld + k0 + ((lane >> 4) << 3);
    V16U a;
    a.h[0] = *reinterpret_cast<const v8h*>(p);
    a.h[1] = *reinterpret_cast<const v8h*>(p + 16);
    return a.v;
}
__device__ __forceinline__ v16h glb_fragB(const _Float16* __restrict__ W,
                                          int ld, int col, int k0, int lane) {
    const _Float16* p = W + (size_t)col * ld + k0 + ((lane >> 4) << 3);
    V16U b;
    b.h[0] = *reinterpret_cast<const v8h*>(p);
    b.h[1] = *reinterpret_cast<const v8h*>(p + 16);
    return b.v;
}

// ---------------------------------------------------------------------------
// WMMA GEMM with TDM->LDS double buffering: C[M,N] = act( X[M,K] @ W[N,K]^T )
// Block = 256 threads = 8 waves; block tile 64x128; wave tile 32x32 (2x2 WMMA)
// ACT: 0 = none, 1 = silu, 2 = sigmoid.  OT: float or _Float16 output.
// NOTE: LDS buffer pointers/offsets are computed arithmetically per use —
// initialized pointer arrays over __shared__ produce addrspacecast static
// initializers that ld.lld rejects.
// ---------------------------------------------------------------------------
template <int ACT, typename OT>
__global__ __launch_bounds__(256)
void gemm_wmma(const _Float16* __restrict__ X, const _Float16* __restrict__ W,
               OT* __restrict__ C, int M, int N, int K) {
    extern __shared__ char smem[];

    const int lane = threadIdx.x & 31;
    const int wave = threadIdx.x >> 5;
    const int mrow0 = (unsigned)blockIdx.y * 64u;     // block M origin
    const int ncol0 = (unsigned)blockIdx.x * 128u;    // block N origin
    const int wm = (wave >> 2) * 32;                  // wave M offset in tile
    const int wn = (wave & 3) * 32;                   // wave N offset in tile

    v8f acc[2][2];
#pragma unroll
    for (int i = 0; i < 2; ++i)
#pragma unroll
        for (int j = 0; j < 2; ++j) acc[i][j] = (v8f)(0.0f);

    // prologue: TDM into buffer 0
    if (threadIdx.x == 0) {
        tdm_load_2d(0u, X, (unsigned)mrow0, 0, K, K, M, 32, 64);
        tdm_load_2d(2u * A_TILE_B, W, (unsigned)ncol0, 0, K, K, N, 32, 128);
        __builtin_amdgcn_s_wait_tensorcnt(0);
    }
    __syncthreads();

    for (int k0 = 0; k0 < K; k0 += 32) {
        const unsigned int buf = (unsigned)(k0 >> 5) & 1u;
        if (k0 + 32 < K && threadIdx.x == 0) {
            const unsigned int nb = buf ^ 1u;
            tdm_load_2d(nb * A_TILE_B, X, (unsigned)mrow0, (unsigned)(k0 + 32),
                        K, K, M, 32, 64);
            tdm_load_2d(2u * A_TILE_B + nb * B_TILE_B, W, (unsigned)ncol0,
                        (unsigned)(k0 + 32), K, K, N, 32, 128);
        }

        const _Float16* bufA = (const _Float16*)(smem + buf * A_TILE_B);
        const _Float16* bufB = (const _Float16*)(smem + 2u * A_TILE_B + buf * B_TILE_B);

        v16h a0 = lds_fragA(bufA, wm + (lane & 15), lane);
        v16h a1 = lds_fragA(bufA, wm + 16 + (lane & 15), lane);
        v16h b0 = lds_fragB(bufB, wn + (lane & 15), lane);
        v16h b1 = lds_fragB(bufB, wn + 16 + (lane & 15), lane);

        acc[0][0] = __builtin_amdgcn_wmma_f32_16x16x32_f16(false, a0, false, b0,
                        (short)0, acc[0][0], false, false);
        acc[0][1] = __builtin_amdgcn_wmma_f32_16x16x32_f16(false, a0, false, b1,
                        (short)0, acc[0][1], false, false);
        acc[1][0] = __builtin_amdgcn_wmma_f32_16x16x32_f16(false, a1, false, b0,
                        (short)0, acc[1][0], false, false);
        acc[1][1] = __builtin_amdgcn_wmma_f32_16x16x32_f16(false, a1, false, b1,
                        (short)0, acc[1][1], false, false);

        if (threadIdx.x == 0) __builtin_amdgcn_s_wait_tensorcnt(0);
        __syncthreads();
    }

    // C/D layout: VGPR r, lane L -> M = r + (L<16?0:8), N = L&15
#pragma unroll
    for (int i = 0; i < 2; ++i) {
        int mb = mrow0 + wm + i * 16 + ((lane >> 4) << 3);
#pragma unroll
        for (int j = 0; j < 2; ++j) {
            int n = ncol0 + wn + j * 16 + (lane & 15);
#pragma unroll
            for (int r = 0; r < 8; ++r) {
                float v = acc[i][j][r];
                if (ACT == 1) v = fsilu(v);
                else if (ACT == 2) v = fsigmoid(v);
                C[(size_t)(mb + r) * N + n] = (OT)v;
            }
        }
    }
}

// ---------------------------------------------------------------------------
// H-projections: beta/fd/sd = sigmoid( X @ Wh[16,K]^T (+ bias) ), N=16 tile.
// Block = 128 threads = 4 waves, each wave one 16-row tile, 3 accumulators.
// ---------------------------------------------------------------------------
__global__ __launch_bounds__(128)
void proj_h_wmma(const _Float16* __restrict__ X,
                 const _Float16* __restrict__ Wb, const _Float16* __restrict__ Wfd,
                 const _Float16* __restrict__ Wsd,
                 const float* __restrict__ bfd, const float* __restrict__ bsd,
                 float* __restrict__ beta, float* __restrict__ fd,
                 float* __restrict__ sd, int K) {
    const int lane = threadIdx.x & 31;
    const int wave = threadIdx.x >> 5;
    const int m0 = blockIdx.x * 64 + wave * 16;

    v8f ab = (v8f)(0.0f), af = (v8f)(0.0f), as = (v8f)(0.0f);

    for (int k0 = 0; k0 < K; k0 += 32) {
        v16h a  = glb_fragA(X, K, m0 + (lane & 15), k0, lane);
        v16h wb = glb_fragB(Wb,  K, lane & 15, k0, lane);
        v16h wf = glb_fragB(Wfd, K, lane & 15, k0, lane);
        v16h ws = glb_fragB(Wsd, K, lane & 15, k0, lane);
        ab = __builtin_amdgcn_wmma_f32_16x16x32_f16(false, a, false, wb,
                 (short)0, ab, false, false);
        af = __builtin_amdgcn_wmma_f32_16x16x32_f16(false, a, false, wf,
                 (short)0, af, false, false);
        as = __builtin_amdgcn_wmma_f32_16x16x32_f16(false, a, false, ws,
                 (short)0, as, false, false);
    }

    int n  = lane & 15;
    int mb = m0 + ((lane >> 4) << 3);
    float biasf = bfd[n], biass = bsd[n];
#pragma unroll
    for (int r = 0; r < 8; ++r) {
        size_t o = (size_t)(mb + r) * H_ + n;
        beta[o] = fsigmoid(ab[r]);
        fd[o]   = fsigmoid(af[r] + biasf);
        sd[o]   = fsigmoid(as[r] + biass);
    }
}

// ---------------------------------------------------------------------------
// Recurrent scan: one block per (b,h), 64 threads, one state element/thread.
// q,k,v,g are f16 (halved HBM traffic); fuses o*g and emits f16 for Wo GEMM.
// ---------------------------------------------------------------------------
__global__ __launch_bounds__(64)
void deltanet_scan(const _Float16* __restrict__ q, const _Float16* __restrict__ k,
                   const _Float16* __restrict__ v, const _Float16* __restrict__ g,
                   const float* __restrict__ beta, const float* __restrict__ fd,
                   const float* __restrict__ sd, _Float16* __restrict__ og) {
    const int b = blockIdx.x / H_;
    const int h = blockIdx.x % H_;
    const int d = threadIdx.x;

    float Sf = 0.0f, Ss = 0.0f;
    const size_t rowbase = (size_t)b * T_;

#pragma unroll 4
    for (int t = 0; t < T_; ++t) {
        const size_t row  = rowbase + t;
        const size_t eidx = row * D_ + h * HD_ + d;
        const size_t sidx = row * H_ + h;

        float qv = (float)q[eidx], kv = (float)k[eidx], vv = (float)v[eidx];
        float bt = beta[sidx], ft = fd[sidx], st = sd[sidx];

        Sf *= ft;
        Ss *= st;
        float o = 0.5f * qv * (Sf + Ss);
        float upd = bt * kv * vv;
        Sf += upd;
        Ss += upd;
        float Sf_new = Sf + MIXC * Ss;
        Ss = Ss + MIXC * Sf;          // pre-mix Sf
        Sf = Sf_new;

        og[eidx] = (_Float16)(o * (float)g[eidx]);
    }
}

// ---------------------------------------------------------------------------
// Host-side orchestration
// ---------------------------------------------------------------------------
static inline size_t align256(size_t x) { return (x + 255) & ~(size_t)255; }

extern "C" void kernel_launch(void* const* d_in, const int* in_sizes, int n_in,
                              void* d_out, int out_size, void* d_ws, size_t ws_size,
                              hipStream_t stream) {
    (void)in_sizes; (void)n_in; (void)out_size; (void)ws_size;

    const float* x   = (const float*)d_in[0];
    const float* Wq  = (const float*)d_in[1];
    const float* Wk  = (const float*)d_in[2];
    const float* Wv  = (const float*)d_in[3];
    const float* Wo  = (const float*)d_in[4];
    const float* Wb  = (const float*)d_in[5];
    const float* Wfd = (const float*)d_in[6];
    const float* bfd = (const float*)d_in[7];
    const float* Wsd = (const float*)d_in[8];
    const float* bsd = (const float*)d_in[9];
    const float* Wg  = (const float*)d_in[10];
    float* out = (float*)d_out;

    char* ws = (char*)d_ws;
    size_t off = 0;
    auto alloc = [&](size_t bytes) { char* p = ws + off; off += align256(bytes); return p; };

    _Float16* hx   = (_Float16*)alloc((size_t)M_ * D_ * 2);
    _Float16* hWq  = (_Float16*)alloc((size_t)D_ * D_ * 2);
    _Float16* hWk  = (_Float16*)alloc((size_t)D_ * D_ * 2);
    _Float16* hWv  = (_Float16*)alloc((size_t)D_ * D_ * 2);
    _Float16* hWo  = (_Float16*)alloc((size_t)D_ * D_ * 2);
    _Float16* hWg  = (_Float16*)alloc((size_t)D_ * D_ * 2);
    _Float16* hWb  = (_Float16*)alloc((size_t)H_ * D_ * 2);
    _Float16* hWfd = (_Float16*)alloc((size_t)H_ * D_ * 2);
    _Float16* hWsd = (_Float16*)alloc((size_t)H_ * D_ * 2);
    _Float16* qf   = (_Float16*)alloc((size_t)M_ * D_ * 2);
    _Float16* kf   = (_Float16*)alloc((size_t)M_ * D_ * 2);
    _Float16* vf   = (_Float16*)alloc((size_t)M_ * D_ * 2);
    _Float16* gf   = (_Float16*)alloc((size_t)M_ * D_ * 2);
    float* bet  = (float*)alloc((size_t)M_ * H_ * 4);
    float* fdb  = (float*)alloc((size_t)M_ * H_ * 4);
    float* sdb  = (float*)alloc((size_t)M_ * H_ * 4);
    _Float16* og16 = (_Float16*)alloc((size_t)M_ * D_ * 2);

    // 1) casts to f16
    {
        int n4 = M_ * D_ / 4;
        cast_f32_to_f16<<<(n4 + 255) / 256, 256, 0, stream>>>(x, hx, n4);
        int w4 = D_ * D_ / 4;
        cast_f32_to_f16<<<(w4 + 255) / 256, 256, 0, stream>>>(Wq, hWq, w4);
        cast_f32_to_f16<<<(w4 + 255) / 256, 256, 0, stream>>>(Wk, hWk, w4);
        cast_f32_to_f16<<<(w4 + 255) / 256, 256, 0, stream>>>(Wv, hWv, w4);
        cast_f32_to_f16<<<(w4 + 255) / 256, 256, 0, stream>>>(Wo, hWo, w4);
        cast_f32_to_f16<<<(w4 + 255) / 256, 256, 0, stream>>>(Wg, hWg, w4);
        int h4 = H_ * D_ / 4;
        cast_f32_to_f16<<<(h4 + 255) / 256, 256, 0, stream>>>(Wb,  hWb,  h4);
        cast_f32_to_f16<<<(h4 + 255) / 256, 256, 0, stream>>>(Wfd, hWfd, h4);
        cast_f32_to_f16<<<(h4 + 255) / 256, 256, 0, stream>>>(Wsd, hWsd, h4);
    }

    // 2) big projections (TDM + WMMA): q,k,v (silu) and g (sigmoid) -> f16
    dim3 ggrid(D_ / 128, M_ / 64);
    gemm_wmma<1, _Float16><<<ggrid, 256, SMEM_B, stream>>>(hx, hWq, qf, M_, D_, D_);
    gemm_wmma<1, _Float16><<<ggrid, 256, SMEM_B, stream>>>(hx, hWk, kf, M_, D_, D_);
    gemm_wmma<1, _Float16><<<ggrid, 256, SMEM_B, stream>>>(hx, hWv, vf, M_, D_, D_);
    gemm_wmma<2, _Float16><<<ggrid, 256, SMEM_B, stream>>>(hx, hWg, gf, M_, D_, D_);

    // 3) H-wide projections (WMMA, N=16): beta, fd, sd
    proj_h_wmma<<<M_ / 64, 128, 0, stream>>>(hx, hWb, hWfd, hWsd, bfd, bsd,
                                             bet, fdb, sdb, D_);

    // 4) recurrent scan, fused with o*g, emits f16
    deltanet_scan<<<B_ * H_, 64, 0, stream>>>(qf, kf, vf, gf, bet, fdb, sdb, og16);

    // 5) output projection: out = og @ Wo^T (TDM + WMMA, f32 out)
    gemm_wmma<0, float><<<ggrid, 256, SMEM_B, stream>>>(og16, hWo, out, M_, D_, D_);
}
